// MultiHeadCrossAttention2D_20444044329013
// MI455X (gfx1250) — matrix operand order repaired and checked by reading, not compiled, not run on hardware
//
#include <hip/hip_runtime.h>
#include <math.h>

// Problem constants (reference: EMBED=HIDDEN=768, HEADS=12, HEAD_DIM=64, B=4, S=2048)
#define B_ 4
#define S_ 2048
#define E_ 768
#define H_ 12
#define D_ 64
#define M_ (B_ * S_) // 8192 rows for all projection GEMMs

typedef _Float16 v8h  __attribute__((ext_vector_type(8)));
typedef _Float16 v16h __attribute__((ext_vector_type(16)));
typedef float    v8f  __attribute__((ext_vector_type(8)));
typedef int      v4i  __attribute__((ext_vector_type(4)));

#if __has_builtin(__builtin_amdgcn_global_load_async_to_lds_b128)
#define HAVE_ASYNC_LDS 1
typedef __attribute__((address_space(1))) v4i as1_v4i; // global int4
typedef __attribute__((address_space(3))) v4i as3_v4i; // LDS int4
#else
#define HAVE_ASYNC_LDS 0
#endif

static __device__ __forceinline__ void wait_async_zero() {
#if HAVE_ASYNC_LDS
#if __has_builtin(__builtin_amdgcn_s_wait_asynccnt)
  __builtin_amdgcn_s_wait_asynccnt(0);
#else
  asm volatile("s_wait_asynccnt 0x0" ::: "memory");
#endif
#endif
}

static __device__ __forceinline__ v16h mk16(v8h lo, v8h hi) {
  return __builtin_shufflevector(lo, hi, 0,1,2,3,4,5,6,7,8,9,10,11,12,13,14,15);
}
// D = A(16x32 f16) * B(32x16 f16) + C(16x16 f32)
static __device__ __forceinline__ v8f wmma16(v16h a, v16h b, v8f c) {
  return __builtin_amdgcn_wmma_f32_16x16x32_f16(false, a, false, b, (short)0, c,
                                                false, false);
}
static __device__ __forceinline__ v8f vzero8() {
  v8f z;
#pragma unroll
  for (int i = 0; i < 8; ++i) z[i] = 0.0f;
  return z;
}

// ---------------------------------------------------------------------------
// f32 -> f16 conversion (element counts are multiples of 4)
// ---------------------------------------------------------------------------
__global__ __launch_bounds__(256)
void cvt_f32_f16(const float* __restrict__ s, _Float16* __restrict__ d, int n) {
  int i = (blockIdx.x * blockDim.x + threadIdx.x) * 4;
  if (i + 3 < n) {
    float4 v = *(const float4*)(s + i);
    d[i + 0] = (_Float16)v.x;
    d[i + 1] = (_Float16)v.y;
    d[i + 2] = (_Float16)v.z;
    d[i + 3] = (_Float16)v.w;
  }
}

// ---------------------------------------------------------------------------
// WMMA GEMM: C[M=8192, N=768] = A[M,768] @ W[N,768]^T  (+ bias)
// Block = 128 threads (4 waves), block tile 128x128, wave tile 64x64, K step 32.
// Tiles are staged with GLOBAL_LOAD_ASYNC_TO_LDS_B128 (ASYNCcnt) when the
// toolchain exposes the builtin; otherwise a pipelined load/store fallback.
// LDS row stride 48 halfwords keeps all b128 fragment loads 16B aligned.
// mode 0: Qh[m,n]  = f16((acc+bias) * outScale)      (outScale = 1/sqrt(768))
// mode 1: Kh[m,n]  = f16(acc+bias)
// mode 2: Vt[b,h,d,s] = f16(acc+bias)                (head-transposed V)
// mode 3: out[m,n] = alpha*attnO[m,n] + beta*(acc+bias)
// ---------------------------------------------------------------------------
__global__ __launch_bounds__(128)
void gemm_wmma(const _Float16* __restrict__ A, const _Float16* __restrict__ W,
               const float* __restrict__ bias, int mode, float outScale,
               _Float16* __restrict__ outH, const float* __restrict__ attnO,
               const float* __restrict__ alphaP, const float* __restrict__ betaP,
               float* __restrict__ outF) {
  __shared__ _Float16 sA[128 * 48];
  __shared__ _Float16 sB[128 * 48];

  const int t    = threadIdx.x;
  const int lane = t & 31;
  const int wave = t >> 5;
  const int half = lane >> 4;   // which 16-lane half of the wave
  const int lm   = lane & 15;
  const int wm   = wave >> 1;   // 2x2 wave grid -> 128x128 block tile
  const int wn   = wave & 1;
  const int m0   = blockIdx.x * 128;
  const int n0   = blockIdx.y * 128;

  v8f acc[4][4];
#pragma unroll
  for (int mt = 0; mt < 4; ++mt)
#pragma unroll
    for (int nt = 0; nt < 4; ++nt) acc[mt][nt] = vzero8();

  for (int kb = 0; kb < E_; kb += 32) {
    // ---- Stage 128x32 tiles of A and W into LDS ----
#if HAVE_ASYNC_LDS
#pragma unroll
    for (int i = 0; i < 4; ++i) {
      int chunk = i * 128 + t;        // 512 chunks of 8 halfwords per tile
      int row   = chunk >> 2;
      int cc    = chunk & 3;
      __builtin_amdgcn_global_load_async_to_lds_b128(
          (as1_v4i*)&A[(size_t)(m0 + row) * E_ + kb + cc * 8],
          (as3_v4i*)&sA[row * 48 + cc * 8], 0, 0);
      __builtin_amdgcn_global_load_async_to_lds_b128(
          (as1_v4i*)&W[(size_t)(n0 + row) * E_ + kb + cc * 8],
          (as3_v4i*)&sB[row * 48 + cc * 8], 0, 0);
    }
    wait_async_zero();
#else
    v8h ra[4], rb[4];
#pragma unroll
    for (int i = 0; i < 4; ++i) {
      int chunk = i * 128 + t;
      int row   = chunk >> 2;
      int cc    = chunk & 3;
      ra[i] = *(const v8h*)&A[(size_t)(m0 + row) * E_ + kb + cc * 8];
      rb[i] = *(const v8h*)&W[(size_t)(n0 + row) * E_ + kb + cc * 8];
    }
#pragma unroll
    for (int i = 0; i < 4; ++i) {
      int chunk = i * 128 + t;
      int row   = chunk >> 2;
      int cc    = chunk & 3;
      *(v8h*)&sA[row * 48 + cc * 8] = ra[i];
      *(v8h*)&sB[row * 48 + cc * 8] = rb[i];
    }
#endif
    if (kb + 32 < E_) { // pull next K tile toward the caches (global_prefetch)
      __builtin_prefetch(&A[(size_t)(m0 + (t & 127)) * E_ + kb + 32], 0, 1);
      __builtin_prefetch(&W[(size_t)(n0 + (t & 127)) * E_ + kb + 32], 0, 1);
    }
    __syncthreads();

    // B fragments: lane lm = column, K = 16*half .. 16*half+15 (contiguous).
    v16h bf[4];
#pragma unroll
    for (int nt = 0; nt < 4; ++nt) {
      const _Float16* p = &sB[(wn * 64 + nt * 16 + lm) * 48 + 16 * half];
      bf[nt] = mk16(*(const v8h*)p, *(const v8h*)(p + 8));
    }
    // A fragments: lane lm = row, K = 16*half .. 16*half+15 (contiguous).
#pragma unroll
    for (int mt = 0; mt < 4; ++mt) {
      const _Float16* p = &sA[(wm * 64 + mt * 16 + lm) * 48 + 16 * half];
      v16h af = mk16(*(const v8h*)p, *(const v8h*)(p + 8));
#pragma unroll
      for (int nt = 0; nt < 4; ++nt)
        acc[mt][nt] = wmma16(af, bf[nt], acc[mt][nt]);
    }
    __syncthreads();
  }

  float alpha = 0.0f, beta = 0.0f;
  if (mode == 3) { alpha = *alphaP; beta = *betaP; }

#pragma unroll
  for (int nt = 0; nt < 4; ++nt) {
    const int n_g = n0 + wn * 64 + nt * 16 + lm;
    const float bv = bias[n_g];
#pragma unroll
    for (int mt = 0; mt < 4; ++mt) {
#pragma unroll
      for (int r = 0; r < 8; ++r) {
        const int m_g = m0 + wm * 64 + mt * 16 + r + 8 * half; // C layout row
        const float v = acc[mt][nt][r] + bv;
        if (mode <= 1) {
          outH[(size_t)m_g * E_ + n_g] = (_Float16)(v * outScale);
        } else if (mode == 2) {
          const int h = n_g >> 6, d = n_g & 63;
          const int b = m_g >> 11, s = m_g & (S_ - 1);
          outH[(((size_t)b * H_ + h) * D_ + d) * S_ + s] = (_Float16)v;
        } else {
          const size_t idx = (size_t)m_g * E_ + n_g;
          outF[idx] = alpha * attnO[idx] + beta * v;
        }
      }
    }
  }
}

// ---------------------------------------------------------------------------
// Flash-style attention. One wave (32 threads) handles 16 query rows for one
// (b,h). Online softmax; K/V streamed in 32-key chunks; P goes C-layout ->
// LDS -> A-fragment layout. Qh already carries the 1/sqrt(768) scale.
//   Qh,Kh: [B,S,H*D] f16 ; Vt: [B,H,D,S] f16 ; O: [B,S,H*D] f32
// ---------------------------------------------------------------------------
__global__ __launch_bounds__(32)
void attn_wmma(const _Float16* __restrict__ Qh, const _Float16* __restrict__ Kh,
               const _Float16* __restrict__ Vt, float* __restrict__ O) {
  __shared__ _Float16 sP[16 * 48];

  const int bh   = blockIdx.y;           // 0..47
  const int b    = bh / H_;
  const int h    = bh % H_;
  const int q0   = blockIdx.x * 16;      // query-row tile
  const int lane = threadIdx.x;
  const int half = lane >> 4;
  const int lm   = lane & 15;

  // Q A-fragments (K dim = 64 -> two 16x32 fragments), loaded once.
  const _Float16* qrow = Qh + ((size_t)b * S_ + q0 + lm) * E_ + h * D_;
  v16h qf[2];
#pragma unroll
  for (int kk = 0; kk < 2; ++kk) {
    const _Float16* p = qrow + kk * 32 + 16 * half;
    qf[kk] = mk16(*(const v8h*)p, *(const v8h*)(p + 8));
  }

  float m8[8], l8[8];
  v8f accO[4];
#pragma unroll
  for (int r = 0; r < 8; ++r) { m8[r] = -1.0e30f; l8[r] = 0.0f; }
#pragma unroll
  for (int nt = 0; nt < 4; ++nt) accO[nt] = vzero8();

  for (int kv0 = 0; kv0 < S_; kv0 += 32) {
    // ---- S = Q @ K^T over a 16x32 score tile (2 col sub-tiles) ----
    const _Float16* kbase = Kh + ((size_t)b * S_ + kv0) * E_ + h * D_;
    v8f sc[2];
#pragma unroll
    for (int nt = 0; nt < 2; ++nt) {
      v8f s = vzero8();
#pragma unroll
      for (int kk = 0; kk < 2; ++kk) {
        const _Float16* kp = kbase + (size_t)(nt * 16 + lm) * E_ + kk * 32 + 16 * half;
        v16h kf = mk16(*(const v8h*)kp, *(const v8h*)(kp + 8));
        s = wmma16(qf[kk], kf, s);
      }
      sc[nt] = s;
    }

    // ---- online softmax on the 16x32 tile ----
    float scale_r[8];
#pragma unroll
    for (int r = 0; r < 8; ++r) {
      float tmx = fmaxf(sc[0][r], sc[1][r]);
      tmx = fmaxf(tmx, __shfl_xor(tmx, 1, 32));
      tmx = fmaxf(tmx, __shfl_xor(tmx, 2, 32));
      tmx = fmaxf(tmx, __shfl_xor(tmx, 4, 32));
      tmx = fmaxf(tmx, __shfl_xor(tmx, 8, 32));
      const float mn = fmaxf(m8[r], tmx);
      const float scl = __expf(m8[r] - mn);
      const float p0 = __expf(sc[0][r] - mn);
      const float p1 = __expf(sc[1][r] - mn);
      float u = p0 + p1;
      u += __shfl_xor(u, 1, 32);
      u += __shfl_xor(u, 2, 32);
      u += __shfl_xor(u, 4, 32);
      u += __shfl_xor(u, 8, 32);
      l8[r] = l8[r] * scl + u;
      m8[r] = mn;
      scale_r[r] = scl;
      // C-layout (row = r + 8*half, col = lane%16 + 16*nt) -> LDS
      sP[(r + 8 * half) * 48 + lm]      = (_Float16)p0;
      sP[(r + 8 * half) * 48 + 16 + lm] = (_Float16)p1;
    }
    __syncthreads();

#pragma unroll
    for (int nt = 0; nt < 4; ++nt)
#pragma unroll
      for (int r = 0; r < 8; ++r) accO[nt][r] *= scale_r[r];

    // P as a 16x32 A-fragment from LDS
    {
      const _Float16* pp = &sP[lm * 48 + 16 * half];
      v16h pf = mk16(*(const v8h*)pp, *(const v8h*)(pp + 8));
      // V B-fragments: Vt[b,h,d,s] -> column d, contiguous 16 keys per lane.
      const _Float16* vb = Vt + ((size_t)bh * D_) * S_ + kv0;
#pragma unroll
      for (int nt = 0; nt < 4; ++nt) {
        const _Float16* vp = vb + (size_t)(nt * 16 + lm) * S_ + 16 * half;
        v16h vf = mk16(*(const v8h*)vp, *(const v8h*)(vp + 8));
        accO[nt] = wmma16(pf, vf, accO[nt]);
      }
    }
    __syncthreads();
  }

  // ---- finalize: divide by row sums and store f32 ----
#pragma unroll
  for (int r = 0; r < 8; ++r) {
    const float inv = 1.0f / l8[r];
    const int row = q0 + r + 8 * half;
#pragma unroll
    for (int nt = 0; nt < 4; ++nt)
      O[((size_t)b * S_ + row) * E_ + h * D_ + nt * 16 + lm] = accO[nt][r] * inv;
  }
}

// ---------------------------------------------------------------------------
// Host launcher
// ---------------------------------------------------------------------------
extern "C" void kernel_launch(void* const* d_in, const int* in_sizes, int n_in,
                              void* d_out, int out_size, void* d_ws, size_t ws_size,
                              hipStream_t stream) {
  const float* v1    = (const float*)d_in[0];
  const float* v2    = (const float*)d_in[1];
  const float* Wq    = (const float*)d_in[2];
  const float* bq    = (const float*)d_in[3];
  const float* Wk    = (const float*)d_in[4];
  const float* bk    = (const float*)d_in[5];
  const float* Wv    = (const float*)d_in[6];
  const float* bv    = (const float*)d_in[7];
  const float* Wo    = (const float*)d_in[8];
  const float* bo    = (const float*)d_in[9];
  const float* alphaP = (const float*)d_in[10];
  const float* betaP  = (const float*)d_in[11];
  float* out = (float*)d_out;

  const size_t NV = (size_t)M_ * E_;  // 6,291,456
  const size_t NW = (size_t)E_ * E_;  // 589,824

  char* cur = (char*)d_ws;
  auto take = [&](size_t bytes) -> char* {
    char* r = cur;
    cur += (bytes + 255) & ~(size_t)255;
    return r;
  };
  _Float16* v1h  = (_Float16*)take(NV * 2);
  _Float16* v2h  = (_Float16*)take(NV * 2);
  _Float16* wqh  = (_Float16*)take(NW * 2);
  _Float16* wkh  = (_Float16*)take(NW * 2);
  _Float16* wvh  = (_Float16*)take(NW * 2);
  _Float16* woh  = (_Float16*)take(NW * 2);
  _Float16* qh   = (_Float16*)take(NV * 2);
  _Float16* kh   = (_Float16*)take(NV * 2);
  _Float16* vt   = (_Float16*)take(NV * 2);
  float*    of32 = (float*)take(NV * 4);
  // total ~93 MB of d_ws

  auto cvt = [&](const float* s, _Float16* d, size_t n) {
    cvt_f32_f16<<<(unsigned)((n / 4 + 255) / 256), 256, 0, stream>>>(s, d, (int)n);
  };
  cvt(v1, v1h, NV);
  cvt(v2, v2h, NV);
  cvt(Wq, wqh, NW);
  cvt(Wk, wkh, NW);
  cvt(Wv, wvh, NW);
  cvt(Wo, woh, NW);

  const float qscale = 0.03608439182435161f; // 1/sqrt(768)
  dim3 gg(M_ / 128, E_ / 128); // 64 x 6

  // Q = v1 @ Wq.T + bq, pre-scaled
  gemm_wmma<<<gg, 128, 0, stream>>>(v1h, wqh, bq, 0, qscale, qh,
                                    nullptr, nullptr, nullptr, nullptr);
  // K = v2 @ Wk.T + bk
  gemm_wmma<<<gg, 128, 0, stream>>>(v2h, wkh, bk, 1, 1.0f, kh,
                                    nullptr, nullptr, nullptr, nullptr);
  // V = v2 @ Wv.T + bv, stored head-transposed
  gemm_wmma<<<gg, 128, 0, stream>>>(v2h, wvh, bv, 2, 1.0f, vt,
                                    nullptr, nullptr, nullptr, nullptr);

  // attention: softmax(QK^T) V -> of32
  attn_wmma<<<dim3(S_ / 16, B_ * H_), 32, 0, stream>>>(qh, kh, vt, of32);

  // out = alpha*of32 + beta*(v1 @ Wo.T + bo)
  gemm_wmma<<<gg, 128, 0, stream>>>(v1h, woh, bo, 3, 1.0f, nullptr,
                                    of32, alphaP, betaP, out);
}
